// ScaledDotProductAttention_27650999451999
// MI455X (gfx1250) — compile-verified, matrix operand order checked by reading, not run on hardware
//
#include <hip/hip_runtime.h>
#include <hip/hip_bf16.h>
#include <stdint.h>

// CDNA5 (gfx1250) flash-attention, bf16 WMMA with fp32 accumulation.
// B=16, S=2048, D=64, temperature=8.  Mask fills with 1e-9 (value), not -inf.
// - Softmax in base-2 domain (log2e folded into Q fragments)
// - Row sums of P computed by WMMA against an all-ones B matrix
// - Row max via DPP16 ROW_XMASK butterfly (no LDS traffic)
// - Double-buffered K/V staging: one workgroup barrier per 64-key round
// - Two-deep LDS fragment pipelining so ds_load latency overlaps WMMA

typedef __attribute__((ext_vector_type(16))) __bf16 v16bf;
typedef __attribute__((ext_vector_type(8)))  float  v8f;

union Frag { v16bf bf; uint32_t u[8]; };

#define BATCH 16
#define SEQ   2048
#define DIM   64
#define NKB   (SEQ / 64)
#define LOG2E 1.44269504088896340736f
#define QSCALE   (0.125f * LOG2E)       // 1/temperature * log2(e)
#define MASKVAL  (1e-9f * LOG2E)        // masked score in base-2 domain

__device__ __forceinline__ uint16_t f2bf(float x) {
    union { __bf16 h; uint16_t u; } r; r.h = (__bf16)x; return r.u;
}
__device__ __forceinline__ uint32_t pack2bf(float lo, float hi) {
    union { __bf16 h[2]; uint32_t u; } r;
    r.h[0] = (__bf16)lo; r.h[1] = (__bf16)hi;
    return r.u;
}

// DPP16 ROW_XMASK butterfly step: exchange within the 16-lane row.
template<int XMASK>
__device__ __forceinline__ float row_xor(float x) {
    return __int_as_float(__builtin_amdgcn_update_dpp(
        0, __float_as_int(x), 0x160 | XMASK, 0xF, 0xF, true));
}

#define WMMA_BF16(A, B, C) \
    __builtin_amdgcn_wmma_f32_16x16x32_bf16(false, (A).bf, false, (B).bf, (short)0, (C), false, false)

__global__ __launch_bounds__(128)
void attn_wmma_kernel(const float* __restrict__ q, const float* __restrict__ k,
                      const float* __restrict__ v, const unsigned char* __restrict__ mask,
                      float* __restrict__ out)
{
    __shared__ __align__(16) uint16_t sK [2][64 * 64];  // K blocks, row-major bf16 (16 KB)
    __shared__ __align__(16) uint16_t sVt[2][64 * 64];  // V blocks, dim-major bf16 (16 KB)
    __shared__ __align__(16) uint16_t sP [4][16 * 64];  // per-wave P scratch (8 KB)

    const int tid  = threadIdx.x;
    const int wave = tid >> 5;
    const int lane = tid & 31;
    const int n16  = lane & 15;   // column within 16-lane half
    const int h    = lane >> 4;   // half select

    const int b     = blockIdx.x >> 5;
    const int qblk  = blockIdx.x & 31;
    const int qbase = qblk * 64 + wave * 16;

    // K-offset per fragment VGPR slot (16-bit A/B layout, ISA 7.12.2)
    int kofs[8];
#pragma unroll
    for (int vv = 0; vv < 8; ++vv)
        kofs[vv] = 2 * (vv & 3) + 8 * h + 16 * (vv >> 2);

    // ---- Q tile -> two bf16 A fragments, pre-scaled by 1/temp * log2e
    Frag qa0, qa1;
    {
        const float* qrow = q + ((size_t)b * SEQ + qbase + n16) * DIM;
#pragma unroll
        for (int vv = 0; vv < 8; ++vv) {
            const int d0 = kofs[vv];
            qa0.u[vv] = pack2bf(qrow[d0] * QSCALE,      qrow[d0 + 1] * QSCALE);
            qa1.u[vv] = pack2bf(qrow[d0 + 32] * QSCALE, qrow[d0 + 33] * QSCALE);
        }
    }

    // All-ones bf16 B fragment: row-sum of P via WMMA
    Frag bones;
#pragma unroll
    for (int vv = 0; vv < 8; ++vv) bones.u[vv] = 0x3F803F80u;

    v8f o0 = {}, o1 = {}, o2 = {}, o3 = {};   // O accumulators (4 dim slices)
    v8f lacc = {};                            // running softmax denominator
    float mrow[8];
#pragma unroll
    for (int j = 0; j < 8; ++j) mrow[j] = -1e30f;

    const unsigned char* mrowp[8];
#pragma unroll
    for (int j = 0; j < 8; ++j)
        mrowp[j] = mask + ((size_t)b * SEQ + qbase + j + 8 * h) * (size_t)SEQ;

    const size_t kvbatch = (size_t)b * SEQ * DIM;

    // ---- staging helper: 64 keys -> bf16 K (row-major) + V^T (dim-major)
    auto stage = [&](int buf, int blk) {
        const size_t gbase = kvbatch + (size_t)blk * 64 * DIM;
        const float4* k4 = (const float4*)(k + gbase);
        const float4* v4 = (const float4*)(v + gbase);
        uint16_t* dK = sK[buf];
        uint16_t* dV = sVt[buf];
#pragma unroll
        for (int i = 0; i < 8; ++i) {
            const int e4  = i * 512 + tid * 4;
            const int key = e4 >> 6;
            const int dim = e4 & 63;
            const float4 kk  = k4[e4 >> 2];
            const float4 vv4 = v4[e4 >> 2];
            *(uint32_t*)&dK[key * 64 + dim]     = pack2bf(kk.x, kk.y);
            *(uint32_t*)&dK[key * 64 + dim + 2] = pack2bf(kk.z, kk.w);
            dV[(dim + 0) * 64 + key] = f2bf(vv4.x);
            dV[(dim + 1) * 64 + key] = f2bf(vv4.y);
            dV[(dim + 2) * 64 + key] = f2bf(vv4.z);
            dV[(dim + 3) * 64 + key] = f2bf(vv4.w);
        }
    };

    stage(0, 0);
    __syncthreads();

    for (int kb = 0; kb < NKB; ++kb) {
        const int cur = kb & 1;
        // stage next block into the other buffer while computing this one
        if (kb + 1 < NKB) stage(1 - cur, kb + 1);

        const uint16_t* Kc = sK[cur];
        const uint16_t* Vc = sVt[cur];

        // fragment loaders (each lowers to 2x ds_load_b128)
        auto loadK = [&](Frag& f, int t, int half) {
#pragma unroll
            for (int vv = 0; vv < 8; ++vv)
                f.u[vv] = *(const uint32_t*)&Kc[(t * 16 + n16) * 64 + half * 32 + kofs[vv]];
        };
        auto loadV = [&](Frag& f, int slice, int half) {
#pragma unroll
            for (int vv = 0; vv < 8; ++vv)
                f.u[vv] = *(const uint32_t*)&Vc[(slice * 16 + n16) * 64 + half * 32 + kofs[vv]];
        };

        // ---- S = (Q*scale) K^T, four 16-key tiles; two-deep frag pipeline
        v8f c[4];
        {
            Frag fA, fB;
            v8f z = {};
            loadK(fA, 0, 0);
            loadK(fB, 0, 1);
            c[0] = WMMA_BF16(qa0, fA, z);
            loadK(fA, 1, 0);
            c[0] = WMMA_BF16(qa1, fB, c[0]);
            loadK(fB, 1, 1);
            c[1] = WMMA_BF16(qa0, fA, z);
            loadK(fA, 2, 0);
            c[1] = WMMA_BF16(qa1, fB, c[1]);
            loadK(fB, 2, 1);
            c[2] = WMMA_BF16(qa0, fA, z);
            loadK(fA, 3, 0);
            c[2] = WMMA_BF16(qa1, fB, c[2]);
            loadK(fB, 3, 1);
            c[3] = WMMA_BF16(qa0, fA, z);
            c[3] = WMMA_BF16(qa1, fB, c[3]);
        }

        // ---- mask substitution + online softmax (base-2), P -> per-wave LDS
        const int mb = kb * 64 + n16;
        uint16_t* P = sP[wave];
        v8f corrv;
#pragma unroll
        for (int j = 0; j < 8; ++j) {
            float s0 = mrowp[j][mb +  0] ? MASKVAL : c[0][j];
            float s1 = mrowp[j][mb + 16] ? MASKVAL : c[1][j];
            float s2 = mrowp[j][mb + 32] ? MASKVAL : c[2][j];
            float s3 = mrowp[j][mb + 48] ? MASKVAL : c[3][j];

            float t = fmaxf(fmaxf(s0, s1), fmaxf(s2, s3));
            t = fmaxf(t, row_xor<1>(t));
            t = fmaxf(t, row_xor<2>(t));
            t = fmaxf(t, row_xor<4>(t));
            t = fmaxf(t, row_xor<8>(t));

            const float nm = fmaxf(mrow[j], t);
            corrv[j] = __builtin_amdgcn_exp2f(mrow[j] - nm);
            mrow[j] = nm;

            const int r = (j + 8 * h) * 64 + n16;
            P[r +  0] = f2bf(__builtin_amdgcn_exp2f(s0 - nm));
            P[r + 16] = f2bf(__builtin_amdgcn_exp2f(s1 - nm));
            P[r + 32] = f2bf(__builtin_amdgcn_exp2f(s2 - nm));
            P[r + 48] = f2bf(__builtin_amdgcn_exp2f(s3 - nm));
        }
        // vector rescale of running accumulators (packed f32 muls)
        lacc *= corrv;
        o0 *= corrv; o1 *= corrv; o2 *= corrv; o3 *= corrv;

        // per-wave LDS round-trip: HW keeps same-wave DS ops in order; only
        // compiler reordering must be fenced.
        __builtin_amdgcn_wave_barrier();

        // ---- reload P as two A fragments (keys 0-31, 32-63)
        Frag ap0, ap1;
#pragma unroll
        for (int vv = 0; vv < 8; ++vv) {
            ap0.u[vv] = *(const uint32_t*)&P[n16 * 64 + kofs[vv]];
            ap1.u[vv] = *(const uint32_t*)&P[n16 * 64 + 32 + kofs[vv]];
        }

        // ---- O += P*V (4 dim slices x 2 key halves) and l += P*ones,
        //      with two-deep V-fragment pipelining; the lacc WMMAs execute
        //      while the first V fragments are still in flight.
        {
            Frag gA, gB;
            loadV(gA, 0, 0);
            loadV(gB, 0, 1);
            lacc = WMMA_BF16(ap0, bones, lacc);
            lacc = WMMA_BF16(ap1, bones, lacc);
            o0 = WMMA_BF16(ap0, gA, o0);
            loadV(gA, 1, 0);
            o0 = WMMA_BF16(ap1, gB, o0);
            loadV(gB, 1, 1);
            o1 = WMMA_BF16(ap0, gA, o1);
            loadV(gA, 2, 0);
            o1 = WMMA_BF16(ap1, gB, o1);
            loadV(gB, 2, 1);
            o2 = WMMA_BF16(ap0, gA, o2);
            loadV(gA, 3, 0);
            o2 = WMMA_BF16(ap1, gB, o2);
            loadV(gB, 3, 1);
            o3 = WMMA_BF16(ap0, gA, o3);
            o3 = WMMA_BF16(ap1, gB, o3);
        }

        // single barrier per round: publishes next block's staging AND
        // guarantees all waves finished reading buffer `cur` before it is
        // overwritten in round kb+1.
        __syncthreads();
    }

    // ---- epilogue: normalize by row sums (replicated per-lane via WMMA-l)
    float* obase = out + ((size_t)b * SEQ + qbase) * DIM;
#pragma unroll
    for (int j = 0; j < 8; ++j) {
        const float inv_l = 1.0f / lacc[j];
        float* orow = obase + (j + 8 * h) * DIM;
        orow[ 0 + n16] = o0[j] * inv_l;
        orow[16 + n16] = o1[j] * inv_l;
        orow[32 + n16] = o2[j] * inv_l;
        orow[48 + n16] = o3[j] * inv_l;
    }
}

extern "C" void kernel_launch(void* const* d_in, const int* in_sizes, int n_in,
                              void* d_out, int out_size, void* d_ws, size_t ws_size,
                              hipStream_t stream) {
    const float*         q    = (const float*)d_in[0];
    const float*         k    = (const float*)d_in[1];
    const float*         v    = (const float*)d_in[2];
    const unsigned char* mask = (const unsigned char*)d_in[3];
    float*               out  = (float*)d_out;

    dim3 grid(BATCH * (SEQ / 64));   // 512 WGs; each wave owns 16 queries
    dim3 block(128);
    attn_wmma_kernel<<<grid, block, 0, stream>>>(q, k, v, mask, out);
}
